// HybridStateSpaceModel_3521873183086
// MI455X (gfx1250) — compile-verified, matrix operand order checked by reading, not run on hardware
//
#include <hip/hip_runtime.h>

// ---------------------------------------------------------------------------
// Model dims
// ---------------------------------------------------------------------------
#define LAYERS 4
#define BB 4
#define SS 1024
#define DD 1024
#define NSTATE 64
#define HH 16
#define DHH 64
#define DII 2048
#define DFF 4096
#define MTOK (BB * SS)          // 4096 token rows

typedef __attribute__((ext_vector_type(16))) __bf16 bf16x16;
typedef __attribute__((ext_vector_type(8)))  float  f32x8;

// ---------------------------------------------------------------------------
// Helpers
// ---------------------------------------------------------------------------
__device__ __forceinline__ __bf16 f2bf(float f) {
  union { float f; unsigned int u; } in; in.f = f;
  unsigned int r = (in.u + 0x7FFFu + ((in.u >> 16) & 1u)) >> 16;
  union { unsigned short s; __bf16 b; } out; out.s = (unsigned short)r;
  return out.b;
}

union FragU { bf16x16 v; unsigned int u[8]; };

// Gather one 16-bit WMMA operand fragment. rowptr points at the start of the
// logical row (M-row for A operands, N-row for B operands); K is striped over
// the 8 dwords with lo/hi half-wave split per the CDNA5 ISA layout tables.
__device__ __forceinline__ bf16x16 frag_ld_k(const __bf16* rowptr, int hi) {
  FragU f;
#pragma unroll
  for (int d = 0; d < 8; ++d) {
    int k0 = (d < 4) ? (hi * 8 + 2 * d) : (16 + hi * 8 + 2 * (d - 4));
    f.u[d] = *(const unsigned int*)(rowptr + k0);
  }
  return f.v;
}

__device__ __forceinline__ f32x8 wmma_bf16(bf16x16 a, bf16x16 b, f32x8 c) {
  return __builtin_amdgcn_wmma_f32_16x16x32_bf16(false, a, false, b,
                                                 (short)0, c, false, false);
}

__device__ __forceinline__ f32x8 zero8() {
  f32x8 z;
#pragma unroll
  for (int e = 0; e < 8; ++e) z[e] = 0.0f;
  return z;
}

__device__ __forceinline__ float silu_f(float x) {
  return x / (1.0f + __expf(-x));
}

// Async copy 32 bytes (2 x b128) from global to LDS, tracked by ASYNCcnt.
// The instruction offset applies to both the global and the LDS address.
__device__ __forceinline__ void async_cp32B(unsigned lds_byte_off,
                                            const __bf16* gsrc) {
  asm volatile(
      "global_load_async_to_lds_b128 %0, %1, off\n\t"
      "global_load_async_to_lds_b128 %0, %1, off offset:16"
      :: "v"(lds_byte_off), "v"(gsrc)
      : "memory");
}

// ---------------------------------------------------------------------------
// GEMM: C[M,N] = A[M,K] (bf16) x W[N,K]^T (bf16) + bias[N]   (fp32 out)
// Block tile 128x128x32, 256 threads (8 waves), each wave -> 32x64 output.
// Double-buffered LDS staging via GLOBAL_LOAD_ASYNC_TO_LDS_B128 + ASYNCcnt.
// ---------------------------------------------------------------------------
#define BM 128
#define BN 128
#define BK 32
#define STAGE_ELEMS (BM * BK + BN * BK)   // 8192 bf16 per buffer (16 KB)

__global__ __launch_bounds__(256) void gemm_bf16_kernel(
    const __bf16* __restrict__ A, const __bf16* __restrict__ W,
    const float* __restrict__ bias, float* __restrict__ C,
    int M, int N, int K) {
  // Single shared allocation -> byte offsets within it are valid LDS
  // addresses for the async-copy VDST operand.
  __shared__ __align__(16) __bf16 smem[2 * STAGE_ELEMS];

  const int tid  = threadIdx.x;
  const int lane = tid & 31;
  const int w    = tid >> 5;
  const int hi   = (lane >> 4) & 1;
  const int ln   = lane & 15;
  const int wm   = w & 3;   // 0..3 -> 32-row stripe
  const int wn   = w >> 2;  // 0..1 -> 64-col stripe
  const int row0 = blockIdx.y * BM;
  const int col0 = blockIdx.x * BN;

  // staging assignment: thread -> one 32B chunk of A tile + one of W tile
  const int srow = tid >> 1;            // 0..127
  const int skk  = (tid & 1) * 16;      // 0 or 16 elements within K tile

  const int gr    = row0 + srow;
  const int gc    = col0 + srow;
  const int grc   = (gr < M) ? gr : (M - 1);   // clamp keeps EXEC full so the
  const int gcc   = (gc < N) ? gc : (N - 1);   // per-wave async count is exact
  const bool a_oob = (gr >= M);
  const bool b_oob = (gc >= N);

  const unsigned la_elem = (unsigned)(srow * BK + skk);
  const unsigned lb_elem = (unsigned)(BM * BK + srow * BK + skk);

  f32x8 acc[2][4];
#pragma unroll
  for (int mt = 0; mt < 2; ++mt)
#pragma unroll
    for (int nt = 0; nt < 4; ++nt) acc[mt][nt] = zero8();

  auto stage = [&](int buf, int k0) {
    unsigned base = (unsigned)(buf * STAGE_ELEMS) * 2u;
    async_cp32B(base + la_elem * 2u, A + (size_t)grc * K + k0 + skk);
    async_cp32B(base + lb_elem * 2u, W + (size_t)gcc * K + k0 + skk);
  };
  auto zfill = [&](int buf) {
    // overwrite clamped (out-of-range) rows with zeros after the async wait
    if (a_oob) {
      uint4 z = {0, 0, 0, 0};
      uint4* d = (uint4*)(smem + buf * STAGE_ELEMS + srow * BK + skk);
      d[0] = z; d[1] = z;
    }
    if (b_oob) {
      uint4 z = {0, 0, 0, 0};
      uint4* d = (uint4*)(smem + buf * STAGE_ELEMS + BM * BK + srow * BK + skk);
      d[0] = z; d[1] = z;
    }
  };

  const int nk = K / BK;
  stage(0, 0);

  for (int it = 0; it < nk; ++it) {
    const int cur = it & 1;
    if (it + 1 < nk) {
      stage(cur ^ 1, (it + 1) * BK);
      // 4 outstanding per wave = exactly the next stage's copies; async ops
      // complete in order, so ASYNCcnt<=4 means the current stage is done.
      asm volatile("s_wait_asynccnt 0x4" ::: "memory");
    } else {
      asm volatile("s_wait_asynccnt 0x0" ::: "memory");
    }
    zfill(cur);
    __syncthreads();

    const __bf16* Abuf = smem + cur * STAGE_ELEMS;
    const __bf16* Bbuf = Abuf + BM * BK;

    bf16x16 af[2], bfg[4];
#pragma unroll
    for (int mt = 0; mt < 2; ++mt)
      af[mt] = frag_ld_k(Abuf + (wm * 32 + mt * 16 + ln) * BK, hi);
#pragma unroll
    for (int nt = 0; nt < 4; ++nt)
      bfg[nt] = frag_ld_k(Bbuf + (wn * 64 + nt * 16 + ln) * BK, hi);
#pragma unroll
    for (int mt = 0; mt < 2; ++mt)
#pragma unroll
      for (int nt = 0; nt < 4; ++nt)
        acc[mt][nt] = wmma_bf16(af[mt], bfg[nt], acc[mt][nt]);
    __syncthreads();
  }

  // ---- epilogue: C row = e + 8*hi within each 16x16 tile ----
#pragma unroll
  for (int mt = 0; mt < 2; ++mt)
#pragma unroll
    for (int nt = 0; nt < 4; ++nt)
#pragma unroll
      for (int e = 0; e < 8; ++e) {
        int r = row0 + wm * 32 + mt * 16 + e + 8 * hi;
        int c = col0 + wn * 64 + nt * 16 + ln;
        if (r < M && c < N)
          C[(size_t)r * N + c] = acc[mt][nt][e] + bias[c];
      }
}

// ---------------------------------------------------------------------------
// Fused attention: flash-style online softmax, WMMA for QK^T and PV.
// grid = (B*H, S/64), block = 128 (4 waves, 16 q-rows each).
// q/k/v are bf16 [B*H, S, 64]; out is fp32 [B*S, D] with heads merged.
// ---------------------------------------------------------------------------
__global__ __launch_bounds__(128) void attn_kernel(
    const __bf16* __restrict__ q, const __bf16* __restrict__ k,
    const __bf16* __restrict__ v, float* __restrict__ out) {
  __shared__ __align__(16) __bf16 Ks[64 * 64];        // [key][dh]
  __shared__ __align__(16) __bf16 Vt[64 * 64];        // [dh][key] (transposed)
  __shared__ __align__(16) __bf16 Ps[4 * 16 * 64];    // per-wave P tile

  const int bh   = blockIdx.x;
  const int b    = bh >> 4;
  const int h    = bh & 15;
  const int tid  = threadIdx.x;
  const int w    = tid >> 5;
  const int lane = tid & 31;
  const int hi   = (lane >> 4) & 1;
  const int ln   = lane & 15;
  const int q0   = blockIdx.y * 64 + w * 16;

  // Q fragments for this wave's 16 rows (A-layout: row = ln)
  const __bf16* qrow = q + ((size_t)bh * SS + q0 + ln) * DHH;
  bf16x16 qf0 = frag_ld_k(qrow, hi);
  bf16x16 qf1 = frag_ld_k(qrow + 32, hi);

  f32x8 acc[4];
#pragma unroll
  for (int dt = 0; dt < 4; ++dt) acc[dt] = zero8();
  float mrow[8], lrow[8];
#pragma unroll
  for (int e = 0; e < 8; ++e) { mrow[e] = -1e30f; lrow[e] = 0.0f; }

  __bf16* pw = Ps + w * (16 * 64);

  for (int kt = 0; kt < SS / 64; ++kt) {
    // ---- stage K tile [64 keys][64 dh] (vectorized) ----
    {
      int r = tid >> 1, off = (tid & 1) * 32;
      const uint4* src =
          (const uint4*)(k + ((size_t)bh * SS + kt * 64 + r) * DHH + off);
      uint4* dst = (uint4*)(Ks + r * 64 + off);
      dst[0] = src[0]; dst[1] = src[1]; dst[2] = src[2]; dst[3] = src[3];
    }
    // ---- stage V transposed [dh][key] ----
#pragma unroll 4
    for (int i = 0; i < 32; ++i) {
      int flat = tid * 32 + i;
      int key = flat >> 6, dh = flat & 63;
      Vt[dh * 64 + key] = v[((size_t)bh * SS + kt * 64 + key) * DHH + dh];
    }
    __syncthreads();

    // ---- S = scale * Q K^T : 4 key tiles x 2 k-steps ----
    f32x8 sc[4];
#pragma unroll
    for (int nt = 0; nt < 4; ++nt) {
      sc[nt] = zero8();
      bf16x16 b0 = frag_ld_k(Ks + (nt * 16 + ln) * 64, hi);
      bf16x16 b1 = frag_ld_k(Ks + (nt * 16 + ln) * 64 + 32, hi);
      sc[nt] = wmma_bf16(qf0, b0, sc[nt]);
      sc[nt] = wmma_bf16(qf1, b1, sc[nt]);
#pragma unroll
      for (int e = 0; e < 8; ++e) sc[nt][e] *= 0.125f;   // 1/sqrt(64)
    }

    // ---- online softmax update (row = e + 8*hi, cols across 16 lanes) ----
#pragma unroll
    for (int e = 0; e < 8; ++e) {
      float mx = fmaxf(fmaxf(sc[0][e], sc[1][e]), fmaxf(sc[2][e], sc[3][e]));
      mx = fmaxf(mx, __shfl_xor(mx, 1));
      mx = fmaxf(mx, __shfl_xor(mx, 2));
      mx = fmaxf(mx, __shfl_xor(mx, 4));
      mx = fmaxf(mx, __shfl_xor(mx, 8));
      float mnew = fmaxf(mrow[e], mx);
      float corr = __expf(mrow[e] - mnew);
      float rs = 0.0f;
#pragma unroll
      for (int t = 0; t < 4; ++t) {
        float pv = __expf(sc[t][e] - mnew);
        sc[t][e] = pv;
        rs += pv;
      }
      rs += __shfl_xor(rs, 1);
      rs += __shfl_xor(rs, 2);
      rs += __shfl_xor(rs, 4);
      rs += __shfl_xor(rs, 8);
      lrow[e] = lrow[e] * corr + rs;
      mrow[e] = mnew;
#pragma unroll
      for (int dt = 0; dt < 4; ++dt) acc[dt][e] *= corr;
    }

    // ---- write P (bf16) through LDS to re-shape into A-fragment layout ----
#pragma unroll
    for (int t = 0; t < 4; ++t)
#pragma unroll
      for (int e = 0; e < 8; ++e)
        pw[(e + 8 * hi) * 64 + t * 16 + ln] = f2bf(sc[t][e]);

    // ---- O += P V : 4 dh tiles x 2 key k-steps ----
#pragma unroll
    for (int dt = 0; dt < 4; ++dt) {
#pragma unroll
      for (int ks = 0; ks < 2; ++ks) {
        bf16x16 pa = frag_ld_k(pw + ln * 64 + ks * 32, hi);
        bf16x16 vb = frag_ld_k(Vt + (dt * 16 + ln) * 64 + ks * 32, hi);
        acc[dt] = wmma_bf16(pa, vb, acc[dt]);
      }
    }
    __syncthreads();
  }

  // ---- normalize + write merged-head output ----
#pragma unroll
  for (int dt = 0; dt < 4; ++dt)
#pragma unroll
    for (int e = 0; e < 8; ++e) {
      int qi = q0 + e + 8 * hi;
      int dh = dt * 16 + ln;
      out[((size_t)b * SS + qi) * DD + h * DHH + dh] = acc[dt][e] / lrow[e];
    }
}

// ---------------------------------------------------------------------------
// Elementwise / small kernels
// ---------------------------------------------------------------------------
__global__ void cvt_bf16_kernel(const float* __restrict__ in,
                                __bf16* __restrict__ out, size_t n) {
  size_t i = (size_t)blockIdx.x * 256 + threadIdx.x;
  if (i < n) out[i] = f2bf(in[i]);
}

__global__ void copyf_kernel(float* __restrict__ dst,
                             const float* __restrict__ src, size_t n) {
  size_t i = (size_t)blockIdx.x * 256 + threadIdx.x;
  if (i < n) dst[i] = src[i];
}

__global__ void addf_kernel(float* __restrict__ x,
                            const float* __restrict__ y, size_t n) {
  size_t i = (size_t)blockIdx.x * 256 + threadIdx.x;
  if (i < n) x[i] += y[i];
}

// xi = silu(xr[:, :DII]) ; xr is [M, 2*DII]
__global__ void silu_xi_kernel(const float* __restrict__ xr,
                               float* __restrict__ xi, size_t n) {
  size_t i = (size_t)blockIdx.x * 256 + threadIdx.x;
  if (i >= n) return;
  size_t row = i / DII, c = i % DII;
  xi[i] = silu_f(xr[row * (2 * DII) + c]);
}

// y = ssum * xi * silu(res)   (res = xr[:, DII:])
__global__ void ygate_kernel(float* __restrict__ xi,
                             const float* __restrict__ xr,
                             const float* __restrict__ ssum, size_t n) {
  size_t i = (size_t)blockIdx.x * 256 + threadIdx.x;
  if (i >= n) return;
  size_t row = i / DII, c = i % DII;
  float r = xr[row * (2 * DII) + DII + c];
  xi[i] = ssum[row] * xi[i] * silu_f(r);
}

// in-place softplus on [M, D] + per-row mean -> mrow
__global__ __launch_bounds__(256) void softplus_mean_kernel(
    float* __restrict__ d, float* __restrict__ mrow) {
  int row = blockIdx.x, tid = threadIdx.x;
  __shared__ float red[256];
  float s = 0.0f;
  for (int i = tid; i < DD; i += 256) {
    float x = d[(size_t)row * DD + i];
    float sp = (x > 20.0f) ? x : log1pf(__expf(x));
    d[(size_t)row * DD + i] = sp;
    s += sp;
  }
  red[tid] = s;
  __syncthreads();
  for (int off = 128; off > 0; off >>= 1) {
    if (tid < off) red[tid] += red[tid + off];
    __syncthreads();
  }
  if (tid == 0) mrow[row] = red[0] / (float)DD;
}

// A_disc = exp(-exp(A_log[n]) * mrow[row]) ; n over NSTATE
__global__ void adisc_kernel(float* __restrict__ Ad,
                             const float* __restrict__ A_log,
                             const float* __restrict__ mrow, size_t n) {
  size_t i = (size_t)blockIdx.x * 256 + threadIdx.x;
  if (i >= n) return;
  size_t row = i >> 6;
  int nn = (int)(i & 63);
  Ad[i] = __expf(-__expf(A_log[nn]) * mrow[row]);
}

// sequential scan over S; grid = B, block = NSTATE threads
__global__ __launch_bounds__(NSTATE) void scan_kernel(
    const float* __restrict__ Ad, const float* __restrict__ U,
    float* __restrict__ ssum) {
  int b = blockIdx.x, n = threadIdx.x;
  __shared__ float red[NSTATE];
  float state = 0.0f;
  for (int s = 0; s < SS; ++s) {
    size_t idx = ((size_t)(b * SS + s)) * NSTATE + n;
    state = Ad[idx] * state + U[idx];
    red[n] = state;
    __syncthreads();
    for (int off = NSTATE / 2; off > 0; off >>= 1) {
      if (n < off) red[n] += red[n + off];
      __syncthreads();
    }
    if (n == 0) ssum[b * SS + s] = red[0];
    __syncthreads();
  }
}

// x = LN(x + add (+ x*Dp)) with gamma/beta; one block per row
__global__ __launch_bounds__(256) void ln_kernel(
    float* __restrict__ x, const float* __restrict__ add,
    const float* __restrict__ Dp, const float* __restrict__ g,
    const float* __restrict__ be) {
  int row = blockIdx.x, tid = threadIdx.x;
  __shared__ float r1[256], r2[256];
  float s = 0.0f, s2 = 0.0f;
  for (int i = tid; i < DD; i += 256) {
    float xv = x[(size_t)row * DD + i];
    float p = xv + add[(size_t)row * DD + i];
    if (Dp) p += xv * Dp[i];
    s += p;
    s2 += p * p;
  }
  r1[tid] = s;
  r2[tid] = s2;
  __syncthreads();
  for (int off = 128; off > 0; off >>= 1) {
    if (tid < off) { r1[tid] += r1[tid + off]; r2[tid] += r2[tid + off]; }
    __syncthreads();
  }
  float mu = r1[0] / (float)DD;
  float var = r2[0] / (float)DD - mu * mu;
  float rstd = rsqrtf(var + 1e-5f);
  for (int i = tid; i < DD; i += 256) {
    float xv = x[(size_t)row * DD + i];
    float p = xv + add[(size_t)row * DD + i];
    if (Dp) p += xv * Dp[i];
    x[(size_t)row * DD + i] = (p - mu) * rstd * g[i] + be[i];
  }
}

__global__ void gelu_kernel(float* __restrict__ h, size_t n) {
  size_t i = (size_t)blockIdx.x * 256 + threadIdx.x;
  if (i >= n) return;
  float t = h[i];
  h[i] = 0.5f * t * (1.0f + erff(t * 0.70710678f));
}

// split [M, 3*D] qkv into bf16 q/k/v [B*H, S, DH]
__global__ void qkv_split_kernel(const float* __restrict__ qkv,
                                 __bf16* __restrict__ qb,
                                 __bf16* __restrict__ kb,
                                 __bf16* __restrict__ vb, size_t n) {
  size_t i = (size_t)blockIdx.x * 256 + threadIdx.x;
  if (i >= n) return;
  size_t row = i / (3 * DD);
  int c = (int)(i % (3 * DD));
  int p = c / DD;
  int r = c % DD;
  int h = r >> 6, dh = r & 63;
  int b = (int)(row >> 10), s = (int)(row & 1023);
  __bf16 val = f2bf(qkv[i]);
  size_t di = (((size_t)(b * HH + h)) * SS + s) * DHH + dh;
  (p == 0 ? qb : (p == 1 ? kb : vb))[di] = val;
}

// ---------------------------------------------------------------------------
// Host orchestration
// ---------------------------------------------------------------------------
static inline unsigned gblk(size_t n) { return (unsigned)((n + 255) / 256); }

extern "C" void kernel_launch(void* const* d_in, const int* in_sizes, int n_in,
                              void* d_out, int out_size, void* d_ws,
                              size_t ws_size, hipStream_t stream) {
  (void)in_sizes; (void)n_in; (void)out_size; (void)ws_size;

  const float* x_in  = (const float*)d_in[0];
  const float* A_log = (const float*)d_in[1];
  const float* Dp    = (const float*)d_in[2];
  const float* W_in  = (const float*)d_in[3];
  const float* b_in  = (const float*)d_in[4];
  const float* W_xp  = (const float*)d_in[5];
  const float* b_xp  = (const float*)d_in[6];
  const float* W_dt  = (const float*)d_in[7];
  const float* b_dt  = (const float*)d_in[8];
  const float* W_out = (const float*)d_in[9];
  const float* b_out = (const float*)d_in[10];
  const float* W_qkv = (const float*)d_in[11];
  const float* b_qkv = (const float*)d_in[12];
  const float* W_ao  = (const float*)d_in[13];
  const float* b_ao  = (const float*)d_in[14];
  const float* g1    = (const float*)d_in[15];
  const float* be1   = (const float*)d_in[16];
  const float* g2    = (const float*)d_in[17];
  const float* be2   = (const float*)d_in[18];
  const float* W_f1  = (const float*)d_in[19];
  const float* b_f1  = (const float*)d_in[20];
  const float* W_f2  = (const float*)d_in[21];
  const float* b_f2  = (const float*)d_in[22];

  char* p = (char*)d_ws;
  auto alloc = [&](size_t bytes) -> char* {
    char* r = p;
    p += (bytes + 255) & ~(size_t)255;
    return r;
  };

  const int M = MTOK;
  float*  xbuf  = (float*)alloc((size_t)M * DD * 4);         // residual stream
  float*  big0  = (float*)alloc((size_t)M * DFF * 4);        // xr / qkv / h
  float*  act1  = (float*)alloc((size_t)M * DII * 4);        // xi / y
  float*  tmpD  = (float*)alloc((size_t)M * DD * 4);         // [M,D] scratch
  float*  mrow  = (float*)alloc((size_t)M * 4);
  float*  ubuf  = (float*)alloc((size_t)M * NSTATE * 4);
  float*  adbuf = (float*)alloc((size_t)M * NSTATE * 4);
  float*  ssum  = (float*)alloc((size_t)M * 4);
  __bf16* abuf  = (__bf16*)alloc((size_t)M * DFF * 2);       // act bf16
  __bf16* wbuf  = (__bf16*)alloc((size_t)DFF * DD * 2);      // weight bf16
  __bf16* qb    = (__bf16*)alloc((size_t)M * DD * 2);
  __bf16* kb    = (__bf16*)alloc((size_t)M * DD * 2);
  __bf16* vb    = (__bf16*)alloc((size_t)M * DD * 2);

  auto cvt = [&](const float* src, __bf16* dst, size_t n) {
    cvt_bf16_kernel<<<gblk(n), 256, 0, stream>>>(src, dst, n);
  };
  auto gemm = [&](const __bf16* A, const __bf16* W, const float* bias,
                  float* C, int Mm, int Nn, int Kk) {
    dim3 grid((Nn + BN - 1) / BN, (Mm + BM - 1) / BM);
    gemm_bf16_kernel<<<grid, 256, 0, stream>>>(A, W, bias, C, Mm, Nn, Kk);
  };

  copyf_kernel<<<gblk((size_t)M * DD), 256, 0, stream>>>(xbuf, x_in,
                                                         (size_t)M * DD);

  for (int l = 0; l < LAYERS; ++l) {
    // ----------------- Mamba block -----------------
    cvt(xbuf, abuf, (size_t)M * DD);
    cvt(W_in + (size_t)l * 2 * DII * DD, wbuf, (size_t)2 * DII * DD);
    gemm(abuf, wbuf, b_in + (size_t)l * 2 * DII, big0, M, 2 * DII, DD);

    silu_xi_kernel<<<gblk((size_t)M * DII), 256, 0, stream>>>(
        big0, act1, (size_t)M * DII);

    cvt(act1, abuf, (size_t)M * DII);
    cvt(W_dt + (size_t)l * DD * DII, wbuf, (size_t)DD * DII);
    gemm(abuf, wbuf, b_dt + (size_t)l * DD, tmpD, M, DD, DII);
    softplus_mean_kernel<<<M, 256, 0, stream>>>(tmpD, mrow);

    cvt(W_xp + (size_t)l * NSTATE * DII, wbuf, (size_t)NSTATE * DII);
    gemm(abuf, wbuf, b_xp + (size_t)l * NSTATE, ubuf, M, NSTATE, DII);

    adisc_kernel<<<gblk((size_t)M * NSTATE), 256, 0, stream>>>(
        adbuf, A_log + (size_t)l * NSTATE, mrow, (size_t)M * NSTATE);
    scan_kernel<<<BB, NSTATE, 0, stream>>>(adbuf, ubuf, ssum);

    ygate_kernel<<<gblk((size_t)M * DII), 256, 0, stream>>>(
        act1, big0, ssum, (size_t)M * DII);

    cvt(act1, abuf, (size_t)M * DII);
    cvt(W_out + (size_t)l * DD * DII, wbuf, (size_t)DD * DII);
    gemm(abuf, wbuf, b_out + (size_t)l * DD, tmpD, M, DD, DII);

    ln_kernel<<<M, 256, 0, stream>>>(xbuf, tmpD, Dp + (size_t)l * DD,
                                     g1 + (size_t)l * DD,
                                     be1 + (size_t)l * DD);

    // ----------------- Attention block -----------------
    cvt(xbuf, abuf, (size_t)M * DD);
    cvt(W_qkv + (size_t)l * 3 * DD * DD, wbuf, (size_t)3 * DD * DD);
    gemm(abuf, wbuf, b_qkv + (size_t)l * 3 * DD, big0, M, 3 * DD, DD);

    qkv_split_kernel<<<gblk((size_t)M * 3 * DD), 256, 0, stream>>>(
        big0, qb, kb, vb, (size_t)M * 3 * DD);

    attn_kernel<<<dim3(BB * HH, SS / 64), 128, 0, stream>>>(qb, kb, vb, tmpD);

    cvt(tmpD, abuf, (size_t)M * DD);
    cvt(W_ao + (size_t)l * DD * DD, wbuf, (size_t)DD * DD);
    gemm(abuf, wbuf, b_ao + (size_t)l * DD, tmpD, M, DD, DD);

    ln_kernel<<<M, 256, 0, stream>>>(xbuf, tmpD, nullptr,
                                     g2 + (size_t)l * DD,
                                     be2 + (size_t)l * DD);

    // ----------------- FFN block -----------------
    cvt(xbuf, abuf, (size_t)M * DD);
    cvt(W_f1 + (size_t)l * DFF * DD, wbuf, (size_t)DFF * DD);
    gemm(abuf, wbuf, b_f1 + (size_t)l * DFF, big0, M, DFF, DD);

    gelu_kernel<<<gblk((size_t)M * DFF), 256, 0, stream>>>(big0,
                                                           (size_t)M * DFF);

    cvt(big0, abuf, (size_t)M * DFF);
    cvt(W_f2 + (size_t)l * DD * DFF, wbuf, (size_t)DD * DFF);
    gemm(abuf, wbuf, b_f2 + (size_t)l * DD, tmpD, M, DD, DFF);

    addf_kernel<<<gblk((size_t)M * DD), 256, 0, stream>>>(xbuf, tmpD,
                                                          (size_t)M * DD);
  }

  copyf_kernel<<<gblk((size_t)M * DD), 256, 0, stream>>>((float*)d_out, xbuf,
                                                         (size_t)M * DD);
}
